// HardMaskedAttention_89644557402746
// MI455X (gfx1250) — matrix operand order repaired and checked by reading, not compiled, not run on hardware
//
#include <hip/hip_runtime.h>

typedef __bf16 bf16_t;
typedef __bf16 v16bf __attribute__((ext_vector_type(16)));
typedef float  v8f   __attribute__((ext_vector_type(8)));

#define BATCH 4
#define NQ 128
#define SEQ 16384
#define EMB 256
#define NH 8
#define DH 32
#define NSTRIP 8
#define S_STRIP (SEQ / NSTRIP)

#define TRY_ASYNC 1

// Round-half-up f32->bf16 pair pack: 2x v_add + 1x v_perm.
__device__ inline unsigned pack_bf2(float lo, float hi) {
  union { float f; unsigned u; } a, b;
  a.f = lo; b.f = hi;
  return __builtin_amdgcn_perm(b.u + 0x8000u, a.u + 0x8000u, 0x07060302u);
}

__device__ inline bf16_t f2bf(float f) {
  union { float f; unsigned u; } x; x.f = f;
  union { unsigned short s; bf16_t b; } y;
  y.s = (unsigned short)((x.u + 0x8000u) >> 16);
  return y.b;
}

// ---------------------------------------------------------------------------
// f32 -> bf16 bulk convert (8 elements / thread). n8 = elems/8.
// ---------------------------------------------------------------------------
__global__ void cvt_f32_bf16(const float* __restrict__ src,
                             bf16_t* __restrict__ dst, int n8) {
  const int i = blockIdx.x * blockDim.x + threadIdx.x;
  if (i >= n8) return;
  const float4* s = (const float4*)src + (size_t)i * 2;
  const float4 x = s[0], y = s[1];
  unsigned* d = (unsigned*)dst + (size_t)i * 4;
  d[0] = pack_bf2(x.x, x.y);
  d[1] = pack_bf2(x.z, x.w);
  d[2] = pack_bf2(y.x, y.y);
  d[3] = pack_bf2(y.z, y.w);
}

// ---------------------------------------------------------------------------
// Projection GEMM: out[m,n] = (sum_k in[m,k]*W[n,k] + bias[n]) * scale
// A tile (16x256) staged once per block into LDS as bf16; W pre-converted.
// mode 0: bf16 out[b][h][r][d]   mode 1: bf16 out[b][h][d][r]   mode 2: f32 [m,256]
// grid.x = M/16, block = 256 (8 waves, 2 n-tiles each)
// ---------------------------------------------------------------------------
__global__ void proj_gemm(const float* __restrict__ in,
                          const bf16_t* __restrict__ W,
                          const float* __restrict__ bias,
                          float scale, int rowsPerBatch,
                          bf16_t* __restrict__ out_bf,
                          float* __restrict__ out_f32,
                          int mode) {
  __shared__ __align__(16) bf16_t Atile[16 * EMB];   // 8 KB

  const int tid = threadIdx.x;
  const int lane = tid & 31;
  const int wave = tid >> 5;
  const int lm = lane & 15;
  const int half = lane >> 4;
  const int row0 = blockIdx.x * 16;

  // Cooperative stage: each thread converts 16 consecutive f32 -> bf16.
  {
    const int r = tid >> 4;
    const int kc = (tid & 15) * 16;
    const float4* src = (const float4*)(in + (size_t)(row0 + r) * EMB + kc);
    unsigned* dst = (unsigned*)(Atile + r * EMB + kc);
    #pragma unroll
    for (int i = 0; i < 4; ++i) {
      const float4 x = src[i];
      dst[2 * i]     = pack_bf2(x.x, x.y);
      dst[2 * i + 1] = pack_bf2(x.z, x.w);
    }
  }
  __syncthreads();

  v8f acc[2];
  #pragma unroll
  for (int t = 0; t < 2; ++t)
    #pragma unroll
    for (int e = 0; e < 8; ++e) acc[t][e] = 0.0f;

  const bf16_t* arow = Atile + lm * EMB;

  #pragma unroll
  for (int k0 = 0; k0 < EMB; k0 += 32) {
    union { v16bf v; unsigned u[8]; } A;
    const unsigned* pa0 = (const unsigned*)(arow + k0 + half * 16);
    const unsigned* pa1 = (const unsigned*)(arow + k0 + half * 16 + 8);
    #pragma unroll
    for (int i = 0; i < 4; ++i) A.u[i]     = pa0[i];
    #pragma unroll
    for (int i = 0; i < 4; ++i) A.u[4 + i] = pa1[i];

    #pragma unroll
    for (int t = 0; t < 2; ++t) {
      const int n0 = (wave * 2 + t) * 16;
      const unsigned* wp = (const unsigned*)(
          W + (size_t)(n0 + lm) * EMB + k0 + half * 16);
      union { v16bf v; unsigned u[8]; } B;
      #pragma unroll
      for (int i = 0; i < 8; ++i) B.u[i] = wp[i];
      acc[t] = __builtin_amdgcn_wmma_f32_16x16x32_bf16(
          false, A.v, false, B.v, (short)0, acc[t], false, false);
    }
  }

  #pragma unroll
  for (int t = 0; t < 2; ++t) {
    const int n = (wave * 2 + t) * 16 + lm;
    const float bv = bias[n];
    #pragma unroll
    for (int r = 0; r < 8; ++r) {
      const int mm = row0 + r + 8 * half;
      const float v = (acc[t][r] + bv) * scale;
      if (mode == 2) {
        out_f32[(size_t)mm * EMB + n] = v;
      } else {
        const int b = mm / rowsPerBatch;
        const int s = mm % rowsPerBatch;
        const int h = n >> 5;
        const int d = n & 31;
        size_t idx;
        if (mode == 0)
          idx = (((size_t)b * NH + h) * (size_t)rowsPerBatch + s) * DH + d;
        else
          idx = (((size_t)b * NH + h) * DH + d) * (size_t)rowsPerBatch + s;
        out_bf[idx] = f2bf(v);
      }
    }
  }
}

// ---------------------------------------------------------------------------
// Flash attention over one (b,h,S-strip). Block = 256 (8 waves, one 16-query
// tile each). K tiles staged into LDS via async DMA, double-buffered; V and
// mask fetched into registers while the DMA is in flight. Row-sum l kept as
// per-lane partials (corr is row-uniform) and reduced once at strip end.
// ---------------------------------------------------------------------------
__global__ void flash_attn(const bf16_t* __restrict__ qp,   // [B,H,NQ,32]
                           const bf16_t* __restrict__ kp,   // [B,H,S,32]
                           const bf16_t* __restrict__ vpt,  // [B,H,32,S]
                           const float*  __restrict__ mask, // [B,NQ,S]
                           float* __restrict__ partO,       // [B*H,NSTRIP,NQ,32]
                           float* __restrict__ partM,
                           float* __restrict__ partL) {
  __shared__ __align__(16) bf16_t pscr[8][16 * 32];
#if TRY_ASYNC
  __shared__ __align__(16) bf16_t kstage[8][2][32 * DH];   // 2 KB per buffer
#endif

  const int lane = threadIdx.x & 31;
  const int wave = threadIdx.x >> 5;
  const int lm = lane & 15;
  const int half = lane >> 4;
  const int strip = blockIdx.x % NSTRIP;
  const int bh = blockIdx.x / NSTRIP;
  const int b = bh >> 3;
  const int q0 = wave * 16;

  const bf16_t* qbase = qp + (size_t)bh * NQ * DH;
  const bf16_t* kbase = kp + (size_t)bh * SEQ * DH;
  const bf16_t* vbase = vpt + (size_t)bh * DH * SEQ;
  const float*  mbase = mask + (size_t)b * NQ * SEQ;

  // Q A-tile, held in registers for the whole strip
  union { v16bf v; unsigned u[8]; } Aq;
  {
    const bf16_t* qrow = qbase + (size_t)(q0 + lm) * DH;
    const unsigned* r0 = (const unsigned*)(qrow + half * 16);
    const unsigned* r1 = (const unsigned*)(qrow + half * 16 + 8);
    #pragma unroll
    for (int i = 0; i < 4; ++i) Aq.u[i] = r0[i];
    #pragma unroll
    for (int i = 0; i < 4; ++i) Aq.u[4 + i] = r1[i];
  }

  float m_run[8], l_lane[8];   // l_lane: per-lane partial row sums
  v8f O[2];
  #pragma unroll
  for (int r = 0; r < 8; ++r) { m_run[r] = -3.0e38f; l_lane[r] = 0.0f; }
  #pragma unroll
  for (int t = 0; t < 2; ++t)
    #pragma unroll
    for (int e = 0; e < 8; ++e) O[t][e] = 0.0f;

  const int sBeg = strip * S_STRIP;
  const int sEnd = sBeg + S_STRIP;

#if TRY_ASYNC
  auto issueK = [&](int s0, int buf) {
    const char* src = (const char*)(kbase + (size_t)s0 * DH);
    const unsigned lbase = (unsigned)(size_t)(&kstage[wave][buf][0]);
    #pragma unroll
    for (int i = 0; i < 4; ++i) {
      const unsigned loff = lbase + (unsigned)(lane * 16 + i * 512);
      const void* g = src + lane * 16 + i * 512;
      asm volatile("global_load_async_to_lds_b128 %0, %1, off"
                   :: "v"(loff), "v"(g) : "memory");
    }
  };
  issueK(sBeg, 0);
#endif

  int chunk = 0;
  for (int s0 = sBeg; s0 < sEnd; s0 += 32, ++chunk) {
#if TRY_ASYNC
    const int buf = chunk & 1;
    const bool more = (s0 + 32) < sEnd;
    if (more) issueK(s0 + 32, buf ^ 1);
#endif

    // ---- independent fetches first: V tiles + mask rows (overlap K DMA) ----
    union { v16bf v; unsigned u[8]; } Bv[2];
    #pragma unroll
    for (int t = 0; t < 2; ++t) {
      const unsigned* vr = (const unsigned*)(
          vbase + (size_t)(t * 16 + lm) * SEQ + s0 + half * 16);
      #pragma unroll
      for (int i = 0; i < 8; ++i) Bv[t].u[i] = vr[i];
    }
    float mk0[8], mk1[8];
    #pragma unroll
    for (int r = 0; r < 8; ++r) {
      const float* mrow = mbase + (size_t)(q0 + r + 8 * half) * SEQ + s0;
      mk0[r] = mrow[lm];
      mk1[r] = mrow[16 + lm];
    }
    __builtin_prefetch(vbase + (size_t)lm * SEQ + s0 + 32, 0, 0);

#if TRY_ASYNC
    if (more) asm volatile("s_wait_asynccnt 0x4" ::: "memory");
    else      asm volatile("s_wait_asynccnt 0x0" ::: "memory");
#endif

    // ---- scores: two 16x16 tiles, K-dim = dh = 32 (one WMMA each) ----
    v8f Sc[2];
    #pragma unroll
    for (int t = 0; t < 2; ++t) {
#if TRY_ASYNC
      const unsigned* kr = (const unsigned*)(
          &kstage[wave][buf][(t * 16 + lm) * DH + half * 16]);
#else
      const unsigned* kr = (const unsigned*)(
          kbase + (size_t)(s0 + t * 16 + lm) * DH + half * 16);
#endif
      union { v16bf v; unsigned u[8]; } B;
      #pragma unroll
      for (int i = 0; i < 8; ++i) B.u[i] = kr[i];
      v8f z;
      #pragma unroll
      for (int e = 0; e < 8; ++e) z[e] = 0.0f;
      Sc[t] = __builtin_amdgcn_wmma_f32_16x16x32_bf16(
          false, Aq.v, false, B.v, (short)0, z, false, false);
    }

    // ---- mask + online softmax (only the max needs a per-chunk reduce) ----
    float mNew[8];
    #pragma unroll
    for (int r = 0; r < 8; ++r) {
      float x0 = (mk0[r] < 0.5f) ? -__builtin_inff() : Sc[0][r];
      float x1 = (mk1[r] < 0.5f) ? -__builtin_inff() : Sc[1][r];
      Sc[0][r] = x0; Sc[1][r] = x1;
      float mx = fmaxf(x0, x1);
      #pragma unroll
      for (int off = 1; off < 16; off <<= 1)
        mx = fmaxf(mx, __shfl_xor(mx, off, 32));
      mNew[r] = fmaxf(mx, m_run[r]);   // floor at -3e38 -> exp(-inf-finite)=0
    }

    bf16_t* ps = pscr[wave];
    #pragma unroll
    for (int r = 0; r < 8; ++r) {
      const float corr = __expf(m_run[r] - mNew[r]);
      const float p0 = __expf(Sc[0][r] - mNew[r]);
      const float p1 = __expf(Sc[1][r] - mNew[r]);
      l_lane[r] = l_lane[r] * corr + (p0 + p1);   // deferred cross-lane sum
      m_run[r] = mNew[r];
      O[0][r] *= corr;
      O[1][r] *= corr;
      const int mm = r + 8 * half;
      ps[mm * 32 + lm]      = f2bf(p0);
      ps[mm * 32 + 16 + lm] = f2bf(p1);
    }

    // ---- reread P in A layout (wave-private LDS slice) ----
    union { v16bf v; unsigned u[8]; } Ap;
    {
      const unsigned* r0 = (const unsigned*)(ps + lm * 32 + half * 16);
      const unsigned* r1 = (const unsigned*)(ps + lm * 32 + half * 16 + 8);
      #pragma unroll
      for (int i = 0; i < 4; ++i) Ap.u[i] = r0[i];
      #pragma unroll
      for (int i = 0; i < 4; ++i) Ap.u[4 + i] = r1[i];
    }

    // ---- O += P @ V ----
    #pragma unroll
    for (int t = 0; t < 2; ++t) {
      O[t] = __builtin_amdgcn_wmma_f32_16x16x32_bf16(
          false, Ap.v, false, Bv[t].v, (short)0, O[t], false, false);
    }
  }

  // ---- final cross-lane reduction of l (once per strip) + store partials ----
  const size_t pb = (size_t)bh * NSTRIP + strip;
  float* Op = partO + pb * NQ * DH;
  float* Mp = partM + pb * NQ;
  float* Lp = partL + pb * NQ;
  #pragma unroll
  for (int r = 0; r < 8; ++r) {
    float ls = l_lane[r];
    #pragma unroll
    for (int off = 1; off < 16; off <<= 1)
      ls += __shfl_xor(ls, off, 32);
    const int qq = q0 + r + 8 * half;
    Op[(size_t)qq * DH + lm]      = O[0][r];
    Op[(size_t)qq * DH + 16 + lm] = O[1][r];
    if (lm == 0) { Mp[qq] = m_run[r]; Lp[qq] = ls; }
  }
}

// ---------------------------------------------------------------------------
// Combine strips: one thread per (b,h,q,d). Writes ctx f32 [B,NQ,E].
// ---------------------------------------------------------------------------
__global__ void combine_strips(const float* __restrict__ partO,
                               const float* __restrict__ partM,
                               const float* __restrict__ partL,
                               float* __restrict__ ctx) {
  const int idx = blockIdx.x * blockDim.x + threadIdx.x;
  if (idx >= BATCH * NH * NQ * DH) return;
  const int d = idx & 31;
  const int q = (idx >> 5) & (NQ - 1);
  const int bh = idx >> 12;
  const int b = bh >> 3;
  const int h = bh & 7;

  float M = -3.0e38f;
  #pragma unroll
  for (int s = 0; s < NSTRIP; ++s)
    M = fmaxf(M, partM[((size_t)bh * NSTRIP + s) * NQ + q]);

  float num = 0.0f, den = 0.0f;
  #pragma unroll
  for (int s = 0; s < NSTRIP; ++s) {
    const size_t pb = (size_t)bh * NSTRIP + s;
    const float w = __expf(partM[pb * NQ + q] - M);
    num += w * partO[(pb * NQ + q) * DH + d];
    den += w * partL[pb * NQ + q];
  }
  ctx[((size_t)b * NQ + q) * EMB + h * DH + d] = (den > 0.0f) ? num / den : 0.0f;
}

// ---------------------------------------------------------------------------
extern "C" void kernel_launch(void* const* d_in, const int* in_sizes, int n_in,
                              void* d_out, int out_size, void* d_ws, size_t ws_size,
                              hipStream_t stream) {
  const float* query = (const float*)d_in[0];
  const float* key   = (const float*)d_in[1];
  const float* value = (const float*)d_in[2];
  const float* mask  = (const float*)d_in[3];
  const float* W_in  = (const float*)d_in[4];
  const float* b_in  = (const float*)d_in[5];
  const float* W_out = (const float*)d_in[6];
  const float* b_out = (const float*)d_in[7];
  float* out = (float*)d_out;

  char* ws = (char*)d_ws;
  size_t off = 0;
  auto take = [&](size_t bytes) -> void* {
    void* p = ws + off;
    off += (bytes + 255) & ~(size_t)255;
    return p;
  };

  bf16_t* qp   = (bf16_t*)take((size_t)BATCH * NH * NQ * DH * sizeof(bf16_t));
  bf16_t* kp   = (bf16_t*)take((size_t)BATCH * NH * SEQ * DH * sizeof(bf16_t));
  bf16_t* vpt  = (bf16_t*)take((size_t)BATCH * NH * DH * SEQ * sizeof(bf16_t));
  float* partO = (float*)take((size_t)BATCH * NH * NSTRIP * NQ * DH * sizeof(float));
  float* partM = (float*)take((size_t)BATCH * NH * NSTRIP * NQ * sizeof(float));
  float* partL = (float*)take((size_t)BATCH * NH * NSTRIP * NQ * sizeof(float));
  float* ctx   = (float*)take((size_t)BATCH * NQ * EMB * sizeof(float));
  bf16_t* Winb = (bf16_t*)take((size_t)3 * EMB * EMB * sizeof(bf16_t));
  bf16_t* Woutb= (bf16_t*)take((size_t)EMB * EMB * sizeof(bf16_t));

  // Weights -> bf16 (tiny)
  cvt_f32_bf16<<<(3 * EMB * EMB / 8 + 255) / 256, 256, 0, stream>>>(
      W_in, Winb, 3 * EMB * EMB / 8);
  cvt_f32_bf16<<<(EMB * EMB / 8 + 255) / 256, 256, 0, stream>>>(
      W_out, Woutb, EMB * EMB / 8);

  const float qscale = 0.17677669529663687f;  // 1/sqrt(32)

  // K projection -> [B,H,S,32] bf16
  proj_gemm<<<(BATCH * SEQ) / 16, 256, 0, stream>>>(
      key, Winb + (size_t)EMB * EMB, b_in + EMB, 1.0f, SEQ, kp, nullptr, 0);
  // V projection -> transposed [B,H,32,S] bf16
  proj_gemm<<<(BATCH * SEQ) / 16, 256, 0, stream>>>(
      value, Winb + (size_t)2 * EMB * EMB, b_in + 2 * EMB, 1.0f, SEQ, vpt, nullptr, 1);
  // Q projection (pre-scaled) -> [B,H,NQ,32] bf16
  proj_gemm<<<(BATCH * NQ) / 16, 256, 0, stream>>>(
      query, Winb, b_in, qscale, NQ, qp, nullptr, 0);

  // Flash attention, split-S
  flash_attn<<<BATCH * NH * NSTRIP, 256, 0, stream>>>(
      qp, kp, vpt, mask, partO, partM, partL);

  // Combine strips -> ctx f32 [B,NQ,E]
  combine_strips<<<(BATCH * NH * NQ * DH + 255) / 256, 256, 0, stream>>>(
      partO, partM, partL, ctx);

  // Output projection -> d_out f32 [B,NQ,E]
  proj_gemm<<<(BATCH * NQ) / 16, 256, 0, stream>>>(
      ctx, Woutb, b_out, 1.0f, NQ, nullptr, out, 2);
}
